// IDCL_22454089023551
// MI455X (gfx1250) — matrix operand order repaired and checked by reading, not compiled, not run on hardware
//
#include <hip/hip_runtime.h>
#include <math.h>

typedef __attribute__((ext_vector_type(16))) _Float16 v16h;
typedef __attribute__((ext_vector_type(8)))  float    v8f;
typedef __attribute__((ext_vector_type(4)))  float    f4;

#define NB   2048   // batch rows
#define NL   128    // seq length
#define ND   512    // feature dim
#define KNN  15
#define INV_TEMP 10.0f

// ---------------------------------------------------------------- init
__global__ void IDCL_zero_kernel(float* out) {
    if (threadIdx.x == 0 && blockIdx.x == 0) out[0] = 0.0f;
}

// ---------------------------------------------------------------- stage 1
// One block per (tensor, row): mean over L, L2 normalize, emit f16 hi/lo split.
// blockDim = 128 threads, each owns 4 contiguous features. Inputs are read
// exactly once -> non-temporal loads keep the 1 GiB stream out of L2 so the
// f16 split buffers stay resident for the WMMA stage.
__global__ void IDCL_mean_norm_kernel(const float* __restrict__ anchor,
                                      const float* __restrict__ modality,
                                      _Float16* __restrict__ Ahi, _Float16* __restrict__ Alo,
                                      _Float16* __restrict__ Mhi, _Float16* __restrict__ Mlo) {
    const int b     = blockIdx.x & (NB - 1);
    const int which = blockIdx.x >> 11;          // 0 = anchor, 1 = modality
    const float* __restrict__ src = which ? modality : anchor;
    _Float16* __restrict__ hi = which ? Mhi : Ahi;
    _Float16* __restrict__ lo = which ? Mlo : Alo;

    const int tid = threadIdx.x;
    const size_t base = (size_t)b * NL * ND + tid * 4;

    f4 acc = {0.f, 0.f, 0.f, 0.f};
    for (int l = 0; l < NL; ++l) {
        f4 v = __builtin_nontemporal_load((const f4*)(src + base + (size_t)l * ND));
        acc += v;
    }
    acc *= (1.0f / (float)NL);

    __shared__ float red[128];
    red[tid] = acc.x * acc.x + acc.y * acc.y + acc.z * acc.z + acc.w * acc.w;
    __syncthreads();
    for (int st = 64; st > 0; st >>= 1) {
        if (tid < st) red[tid] += red[tid + st];
        __syncthreads();
    }
    const float inv = 1.0f / fmaxf(sqrtf(red[0]), 1e-12f);

    const float v0 = acc.x * inv, v1 = acc.y * inv, v2 = acc.z * inv, v3 = acc.w * inv;
    const int o = b * ND + tid * 4;
    const _Float16 h0 = (_Float16)v0, h1 = (_Float16)v1, h2 = (_Float16)v2, h3 = (_Float16)v3;
    hi[o + 0] = h0; hi[o + 1] = h1; hi[o + 2] = h2; hi[o + 3] = h3;
    lo[o + 0] = (_Float16)(v0 - (float)h0);
    lo[o + 1] = (_Float16)(v1 - (float)h1);
    lo[o + 2] = (_Float16)(v2 - (float)h2);
    lo[o + 3] = (_Float16)(v3 - (float)h3);
}

// ---------------------------------------------------------------- stage 2
// Symmetric Gram S = X * X^T with split-f16 precision recovery:
//   S ~= hi*hi^T + hi*lo^T + lo*hi^T    (error ~ |lo|^2, near-fp32 quality)
// One wave32 per 32x32 upper-triangular region (2x2 register tiling, 4
// independent accumulator chains), mirror-stored for the lower triangle.
__device__ inline v16h IDCL_frag2(const _Float16* p0, const _Float16* p1) {
    union { v16h v; f4 f[2]; } u;
    u.f[0] = *(const f4*)p0;   // 8 f16
    u.f[1] = *(const f4*)p1;   // 8 f16
    return u.v;
}

__device__ inline v8f IDCL_mm3(v16h ah, v16h al, v16h bh, v16h bl, v8f c) {
    c = __builtin_amdgcn_wmma_f32_16x16x32_f16(false, ah, false, bh, (short)0, c, false, false);
    c = __builtin_amdgcn_wmma_f32_16x16x32_f16(false, ah, false, bl, (short)0, c, false, false);
    c = __builtin_amdgcn_wmma_f32_16x16x32_f16(false, al, false, bh, (short)0, c, false, false);
    return c;
}

__device__ inline void IDCL_store_tile(float* __restrict__ S, v8f acc,
                                       int rowbase, int colbase,
                                       int l16, bool hih, bool mirror) {
    const int col = colbase + l16;
    const int mb  = rowbase + (hih ? 8 : 0);
#pragma unroll
    for (int r = 0; r < 8; ++r) {
        S[(size_t)(mb + r) * NB + col] = acc[r];
        if (mirror) S[(size_t)col * NB + (mb + r)] = acc[r];
    }
}

__global__ void IDCL_gram_wmma_kernel(const _Float16* __restrict__ Xhi,
                                      const _Float16* __restrict__ Xlo,
                                      float* __restrict__ S) {
    const int wave = threadIdx.x >> 5;
    const int lane = threadIdx.x & 31;
    const int t    = blockIdx.x * 8 + wave;   // triangular region id, 0..2079

    // Decode upper-triangular (tm, tn) over a 64x64 region grid (tn >= tm).
    // cum(m) = 64m - m(m-1)/2 regions precede row m.
    int tm = (int)floorf((129.0f - sqrtf(129.0f * 129.0f - 8.0f * (float)t)) * 0.5f);
    while (64 * (tm + 1) - ((tm + 1) * tm) / 2 <= t) ++tm;   // fp fixup
    while (64 * tm - (tm * (tm - 1)) / 2 > t) --tm;
    const int tn = tm + (t - (64 * tm - (tm * (tm - 1)) / 2));
    const bool mirror = (tm != tn);

    const int  l16 = lane & 15;
    const bool hih = (lane >= 16);

    const int rm0 = tm * 32, rm1 = rm0 + 16;
    const int cn0 = tn * 32, cn1 = cn0 + 16;

    // Per-lane element offsets matching the CDNA5 16-bit A(16x32)/B(32x16) layouts:
    //   A frag: chunk0 = K[kb + (hih?8:0) .. +7], chunk1 = chunk0 + 16
    //   B frag: chunk0 = K[kb + (hih?8:0) .. +7], chunk1 = chunk0 + 16
    const int aoff0 = (rm0 + l16) * ND + (hih ? 8 : 0);
    const int aoff1 = (rm1 + l16) * ND + (hih ? 8 : 0);
    const int boff0 = (cn0 + l16) * ND + (hih ? 8 : 0);
    const int boff1 = (cn1 + l16) * ND + (hih ? 8 : 0);

    v8f acc00 = {}, acc01 = {}, acc10 = {}, acc11 = {};
#pragma unroll 2
    for (int kb = 0; kb < ND; kb += 32) {
        const v16h a0h = IDCL_frag2(Xhi + aoff0 + kb, Xhi + aoff0 + kb + 16);
        const v16h a0l = IDCL_frag2(Xlo + aoff0 + kb, Xlo + aoff0 + kb + 16);
        const v16h a1h = IDCL_frag2(Xhi + aoff1 + kb, Xhi + aoff1 + kb + 16);
        const v16h a1l = IDCL_frag2(Xlo + aoff1 + kb, Xlo + aoff1 + kb + 16);
        const v16h b0h = IDCL_frag2(Xhi + boff0 + kb, Xhi + boff0 + kb + 16);
        const v16h b0l = IDCL_frag2(Xlo + boff0 + kb, Xlo + boff0 + kb + 16);
        const v16h b1h = IDCL_frag2(Xhi + boff1 + kb, Xhi + boff1 + kb + 16);
        const v16h b1l = IDCL_frag2(Xlo + boff1 + kb, Xlo + boff1 + kb + 16);

        acc00 = IDCL_mm3(a0h, a0l, b0h, b0l, acc00);
        acc01 = IDCL_mm3(a0h, a0l, b1h, b1l, acc01);
        acc10 = IDCL_mm3(a1h, a1l, b0h, b0l, acc10);
        acc11 = IDCL_mm3(a1h, a1l, b1h, b1l, acc11);
    }

    IDCL_store_tile(S, acc00, rm0, cn0, l16, hih, mirror);
    IDCL_store_tile(S, acc01, rm0, cn1, l16, hih, mirror);
    IDCL_store_tile(S, acc10, rm1, cn0, l16, hih, mirror);
    IDCL_store_tile(S, acc11, rm1, cn1, l16, hih, mirror);
}

// ---------------------------------------------------------------- stage 3
// One block (256 threads) per row: softmax stats on Sm, 15x argmax on Sa (KNN),
// accumulate -(log(pos+eps) - log(all+eps))/B into out.
__global__ void IDCL_loss_kernel(const float* __restrict__ Sa,
                                 const float* __restrict__ Sm,
                                 float* __restrict__ out) {
    const int i   = blockIdx.x;
    const int tid = threadIdx.x;
    const int T   = 256;

    __shared__ float row[NB];     // Sa row (KNN workspace)
    __shared__ float red[256];
    __shared__ int   redi[256];

    const float* __restrict__ smrow = Sm + (size_t)i * NB;

    // row max of sim_mod (including diagonal, as in reference)
    float m = -INFINITY;
    for (int j = tid; j < NB; j += T) m = fmaxf(m, smrow[j] * INV_TEMP);
    red[tid] = m; __syncthreads();
    for (int st = 128; st > 0; st >>= 1) {
        if (tid < st) red[tid] = fmaxf(red[tid], red[tid + st]);
        __syncthreads();
    }
    m = red[0]; __syncthreads();

    // all_sum over j != i
    float s = 0.0f;
    for (int j = tid; j < NB; j += T)
        if (j != i) s += expf(smrow[j] * INV_TEMP - m);
    red[tid] = s; __syncthreads();
    for (int st = 128; st > 0; st >>= 1) {
        if (tid < st) red[tid] += red[tid + st];
        __syncthreads();
    }
    const float all_sum = red[0]; __syncthreads();

    // load Sa row, mask diagonal
    for (int j = tid; j < NB; j += T)
        row[j] = (j == i) ? -INFINITY : Sa[(size_t)i * NB + j];
    __syncthreads();

    // 15 iterated block argmaxes -> KNN positives; accumulate pos_sum from Sm
    __shared__ float pos_acc;
    if (tid == 0) pos_acc = 0.0f;
    __syncthreads();
    for (int it = 0; it < KNN; ++it) {
        float lm = -INFINITY; int li = -1;
        for (int j = tid; j < NB; j += T) {
            float v = row[j];
            if (v > lm) { lm = v; li = j; }
        }
        red[tid] = lm; redi[tid] = li; __syncthreads();
        for (int st = 128; st > 0; st >>= 1) {
            if (tid < st && red[tid + st] > red[tid]) {
                red[tid] = red[tid + st]; redi[tid] = redi[tid + st];
            }
            __syncthreads();
        }
        const int idx = redi[0];
        __syncthreads();
        if (tid == 0) {
            pos_acc += expf(smrow[idx] * INV_TEMP - m);
            row[idx] = -INFINITY;
        }
        __syncthreads();
    }

    if (tid == 0) {
        float li = logf(pos_acc + 1e-8f) - logf(all_sum + 1e-8f);
        atomicAdd(out, -li / (float)NB);
    }
}

// ---------------------------------------------------------------- launch
extern "C" void kernel_launch(void* const* d_in, const int* in_sizes, int n_in,
                              void* d_out, int out_size, void* d_ws, size_t ws_size,
                              hipStream_t stream) {
    const float* anchor   = (const float*)d_in[0];
    const float* modality = (const float*)d_in[1];
    float* out = (float*)d_out;

    char* ws = (char*)d_ws;
    const size_t HBYTES = (size_t)NB * ND * sizeof(_Float16);   // 2 MB
    _Float16* Ahi = (_Float16*)(ws + 0 * HBYTES);
    _Float16* Alo = (_Float16*)(ws + 1 * HBYTES);
    _Float16* Mhi = (_Float16*)(ws + 2 * HBYTES);
    _Float16* Mlo = (_Float16*)(ws + 3 * HBYTES);
    float* Sa = (float*)(ws + 4 * HBYTES);                       // 16 MB
    float* Sm = (float*)(ws + 4 * HBYTES + (size_t)NB * NB * sizeof(float));

    IDCL_zero_kernel<<<1, 32, 0, stream>>>(out);

    IDCL_mean_norm_kernel<<<2 * NB, 128, 0, stream>>>(anchor, modality,
                                                      Ahi, Alo, Mhi, Mlo);

    // 64x64 region grid, upper triangle: 64*65/2 = 2080 regions, 8 waves/block
    const int nregions = (NB / 32) * ((NB / 32) + 1) / 2;        // 2080
    IDCL_gram_wmma_kernel<<<nregions / 8, 256, 0, stream>>>(Ahi, Alo, Sa);
    IDCL_gram_wmma_kernel<<<nregions / 8, 256, 0, stream>>>(Mhi, Mlo, Sm);

    IDCL_loss_kernel<<<NB, 256, 0, stream>>>(Sa, Sm, out);
}
